// SGNP_14130442404162
// MI455X (gfx1250) — matrix operand order repaired and checked by reading, not compiled, not run on hardware
//
#include <hip/hip_runtime.h>
#include <cmath>

// ---------------- problem constants (match reference) ----------------
#define B_     8
#define NC_    2048
#define NT_    2048
#define NCTX   (B_*NC_)          // 16384 context nodes
#define NTEST  (B_*NT_)          // 16384 test nodes
#define NNODE  (NCTX + NTEST)    // 32768 nodes
#define H_     64
#define KNN_   32
#define NRBF_  16
#define NBLK_  6
#define RBF_GAMMA_ 10.0f

// Async global->LDS path (gfx1250): use if the toolchain exposes the builtins.
#if defined(__has_builtin)
#if __has_builtin(__builtin_amdgcn_global_load_async_to_lds_b128) && \
    __has_builtin(__builtin_amdgcn_s_wait_asynccnt)
#define USE_ASYNC_LDS 1
#endif
#endif
#ifndef USE_ASYNC_LDS
#define USE_ASYNC_LDS 0
#endif

typedef __attribute__((ext_vector_type(16))) _Float16 v16h;
typedef __attribute__((ext_vector_type(8)))  _Float16 v8h;
typedef __attribute__((ext_vector_type(8)))  float    v8f;
// Match the builtin's parameter type exactly: gcc-style 16B int vector.
typedef int v4i_b128 __attribute__((vector_size(16)));

__device__ __forceinline__ float gelu_f(float x) {
  // jax.nn.gelu default (approximate=True, tanh form)
  float x3 = x * x * x;
  return 0.5f * x * (1.0f + tanhf(0.79788456080286535f * (x + 0.044715f * x3)));
}

// ---------------------------------------------------------------------
// Pack an f32 weight matrix W[Kd][N] (row-major) into f16 WMMA B-fragments.
// Fragment layout (per 32x16 K-tile): for lane L, 16 contiguous halves,
// element i = W[kt*32 + (L>>4)*8 + (i&7) + (i>>3)*16][nt*16 + (L&15)].
// Packed offset: ((kt*ntiles_n + nt)*32 + lane)*16 + i  (same total size as W)
// ---------------------------------------------------------------------
__global__ void pack_w_kernel(const float* __restrict__ W, _Float16* __restrict__ out,
                              int Kd, int N) {
  int tid = blockIdx.x * 256 + threadIdx.x;
  int total = Kd * N;
  if (tid >= total) return;
  int i    = tid & 15;
  int lane = (tid >> 4) & 31;
  int frag = tid >> 9;
  int ntn  = N >> 4;
  int kt   = frag / ntn;
  int nt   = frag - kt * ntn;
  int k    = kt * 32 + ((lane >> 4) * 8) + (i & 7) + ((i >> 3) * 16);
  int col  = nt * 16 + (lane & 15);
  out[tid] = (_Float16)W[(size_t)k * N + col];
}

// ---------------------------------------------------------------------
// WMMA GEMM v2: out[M][N] = act(A[M][Kd] * Bpacked + bias), Kd = NKT*32.
// Block = 128 threads = 4 waves. Block covers 4 M-tiles x 4 N-tiles (64x64).
// The block's B-fragment panel (all K, 4 N-tiles) is staged once into LDS
// via async global->LDS (ASYNCcnt) when available, then each wave reuses one
// A fragment across 4 WMMAs (4 f32 accumulators).
// Requires: M % 64 == 0, N % 64 == 0 (true for all layers here).
// ---------------------------------------------------------------------
template <int NKT, int ACT>
__global__ __launch_bounds__(128) void gemm_wmma_v2(
    const _Float16* __restrict__ A,
    const _Float16* __restrict__ Bp,
    const float* __restrict__ bias,
    _Float16* __restrict__ outH,
    float* __restrict__ outF,
    int M, int N) {
  constexpr int NTW = 4;               // N-tiles per block
  constexpr int Kd  = NKT * 32;
  // LDS panel: [kt][ntl][lane][16] halves
  __shared__ __align__(16) _Float16 ldsB[NKT * NTW * 32 * 16];

  const int tid  = threadIdx.x;
  const int lane = tid & 31;
  const int w    = tid >> 5;           // wave id in block (0..3)
  const int ntn  = N >> 4;
  const int nGroups = ntn / NTW;
  const int mg = blockIdx.x / nGroups;
  const int ng = blockIdx.x - mg * nGroups;
  const int ntBase = ng * NTW;

  // ---- stage the B panel into LDS: NKT * 2048 halves, 256 b128 chunks/kt ----
  {
    const int chunks = NKT * 256;      // 16B chunks
#if USE_ASYNC_LDS
    for (int idx = tid; idx < chunks; idx += 128) {
      int kt = idx >> 8;
      int r  = idx & 255;
      const _Float16* src = Bp + (size_t)(kt * ntn + ntBase) * 512 + r * 8;
      _Float16* dst = &ldsB[kt * 2048 + r * 8];
      __builtin_amdgcn_global_load_async_to_lds_b128(
          (__attribute__((address_space(1))) v4i_b128*)src,
          (__attribute__((address_space(3))) v4i_b128*)dst, 0, 0);
    }
    __builtin_amdgcn_s_wait_asynccnt(0);
#else
    for (int idx = tid; idx < chunks; idx += 128) {
      int kt = idx >> 8;
      int r  = idx & 255;
      const uint4* src =
          (const uint4*)(Bp + (size_t)(kt * ntn + ntBase) * 512) + r;
      ((uint4*)&ldsB[kt * 2048])[r] = *src;
    }
#endif
  }
  __syncthreads();

  const int mt   = (mg << 2) + w;
  const int hi   = lane >> 4;
  const int rowA = mt * 16 + (lane & 15);

  v8f c[NTW];
#pragma unroll
  for (int t = 0; t < NTW; ++t)
#pragma unroll
    for (int r = 0; r < 8; ++r) c[t][r] = 0.0f;

  const _Float16* Arow = A + (size_t)rowA * Kd + hi * 8;

#pragma unroll
  for (int kt = 0; kt < NKT; ++kt) {
    // A fragment (ISA 16-bit A 16x32 layout): halves [0..7]=Kbase, [8..15]=Kbase+16
    const v8h* ap = (const v8h*)(Arow + kt * 32);
    v8h a0 = ap[0];
    v8h a1 = ap[2];
    if (kt + 1 < NKT)
      __builtin_prefetch(Arow + (kt + 1) * 32, 0, 1);
    v16h a;
#pragma unroll
    for (int i = 0; i < 8; ++i) { a[i] = a0[i]; a[i + 8] = a1[i]; }
    // One A fragment feeds 4 WMMAs against the LDS-resident B panel
#pragma unroll
    for (int ntl = 0; ntl < NTW; ++ntl) {
      const v8h* bp = (const v8h*)&ldsB[((kt * NTW + ntl) * 32 + lane) * 16];
      v8h b0 = bp[0];
      v8h b1 = bp[1];
      v16h b;
#pragma unroll
      for (int i = 0; i < 8; ++i) { b[i] = b0[i]; b[i + 8] = b1[i]; }
      c[ntl] = __builtin_amdgcn_wmma_f32_16x16x32_f16(false, a, false, b,
                                                      (short)0, c[ntl],
                                                      false, false);
    }
  }

#pragma unroll
  for (int ntl = 0; ntl < NTW; ++ntl) {
    const int col = (ntBase + ntl) * 16 + (lane & 15);
    const float bv = bias ? bias[col] : 0.0f;
#pragma unroll
    for (int r = 0; r < 8; ++r) {
      int row = mt * 16 + hi * 8 + r;  // C/D: VGPR r -> M=r (lanes 0-15), M=8+r (16-31)
      float v = c[ntl][r] + bv;
      if (ACT == 1) v = gelu_f(v);
      size_t off = (size_t)row * N + col;
      if (outH) outH[off] = (_Float16)v;
      if (outF) outF[off] = v;
    }
  }
}

// ---------------------------------------------------------------------
// Layer 0 of embed MLP (K=7, too small for WMMA): builds the 7-dim input
// [obs_embed(4), coords(2), f(1 or 0)] and computes gelu(x@W0 + b0) -> f16.
// ---------------------------------------------------------------------
__global__ void embed_l0_kernel(const float* __restrict__ s_ctx,
                                const float* __restrict__ f_ctx,
                                const float* __restrict__ s_test,
                                const float* __restrict__ obs_table,
                                const float* __restrict__ w0,
                                const float* __restrict__ b0,
                                _Float16* __restrict__ h0) {
  int t = blockIdx.x * 256 + threadIdx.x;   // NNODE*256 threads
  int j = t & 255;
  int n = t >> 8;
  if (n >= NNODE) return;
  float x[7];
  if (n < NCTX) {
#pragma unroll
    for (int d = 0; d < 4; ++d) x[d] = obs_table[4 + d];  // observed row (index 1)
    x[4] = s_ctx[(size_t)n * 2 + 0];
    x[5] = s_ctx[(size_t)n * 2 + 1];
    x[6] = f_ctx[n];
  } else {
    int m = n - NCTX;
#pragma unroll
    for (int d = 0; d < 4; ++d) x[d] = obs_table[d];      // unobserved row (index 0)
    x[4] = s_test[(size_t)m * 2 + 0];
    x[5] = s_test[(size_t)m * 2 + 1];
    x[6] = 0.0f;
  }
  float acc = b0[j];
#pragma unroll
  for (int d = 0; d < 7; ++d) acc += x[d] * w0[d * 256 + j];
  h0[(size_t)n * 256 + j] = (_Float16)gelu_f(acc);
}

// ---------------------------------------------------------------------
// LayerNorm (+ optional residual): one thread per node, H=64.
// ---------------------------------------------------------------------
__global__ void residual_ln_kernel(const float* __restrict__ x,
                                   const float* __restrict__ res,
                                   const float* __restrict__ g,
                                   const float* __restrict__ b,
                                   float* __restrict__ nodes,
                                   _Float16* __restrict__ nodesH,
                                   int n) {
  int t = blockIdx.x * 256 + threadIdx.x;
  if (t >= n) return;
  float v[H_];
  float m = 0.0f;
#pragma unroll
  for (int i = 0; i < H_; ++i) {
    float vv = x[(size_t)t * H_ + i];
    if (res) vv += res[(size_t)t * H_ + i];
    v[i] = vv;
    m += vv;
  }
  m *= (1.0f / H_);
  float var = 0.0f;
#pragma unroll
  for (int i = 0; i < H_; ++i) { float d = v[i] - m; var += d * d; }
  var *= (1.0f / H_);
  float inv = rsqrtf(var + 1e-6f);
#pragma unroll
  for (int i = 0; i < H_; ++i) {
    float y = (v[i] - m) * inv * g[i] + b[i];
    nodes[(size_t)t * H_ + i]  = y;
    nodesH[(size_t)t * H_ + i] = (_Float16)y;
  }
}

// ---------------------------------------------------------------------
// Brute-force kNN: one thread per query node, insertion-sorted top-K over
// the 2048 context points of the same batch.
// ---------------------------------------------------------------------
__global__ void knn_kernel(const float* __restrict__ s_ctx,
                           const float* __restrict__ s_test,
                           int* __restrict__ senders,
                           float* __restrict__ dists) {
  int q = blockIdx.x * 256 + threadIdx.x;
  if (q >= NNODE) return;
  int b;
  float qx, qy;
  if (q < NCTX) {
    b = q / NC_;
    qx = s_ctx[(size_t)q * 2 + 0];
    qy = s_ctx[(size_t)q * 2 + 1];
  } else {
    int r = q - NCTX;
    b = r / NT_;
    qx = s_test[(size_t)r * 2 + 0];
    qy = s_test[(size_t)r * 2 + 1];
  }
  float bd[KNN_];
  int   bi[KNN_];
#pragma unroll
  for (int k = 0; k < KNN_; ++k) { bd[k] = 3.4e38f; bi[k] = 0; }
  const float* cb = s_ctx + (size_t)b * NC_ * 2;
  for (int c = 0; c < NC_; ++c) {
    float dx = cb[2 * c + 0] - qx;
    float dy = cb[2 * c + 1] - qy;
    float d2 = dx * dx + dy * dy;
    if (d2 < bd[KNN_ - 1]) {
      int j = KNN_ - 1;
      while (j > 0 && bd[j - 1] > d2) {
        bd[j] = bd[j - 1];
        bi[j] = bi[j - 1];
        --j;
      }
      bd[j] = d2;
      bi[j] = c;
    }
  }
#pragma unroll
  for (int k = 0; k < KNN_; ++k) {
    senders[(size_t)q * KNN_ + k] = b * NC_ + bi[k];
    dists[(size_t)q * KNN_ + k]   = sqrtf(fmaxf(bd[k], 1e-12f));
  }
}

// ---------------------------------------------------------------------
// Per-block edge softmax: one wave32 per node, lane = edge k.
// ---------------------------------------------------------------------
__global__ void edge_softmax_kernel(const float* __restrict__ dists,
                                    const float* __restrict__ centers,
                                    const float* __restrict__ rw,
                                    const float* __restrict__ rbp,
                                    float* __restrict__ edgew) {
  int node = blockIdx.x * (blockDim.x >> 5) + (threadIdx.x >> 5);
  int lane = threadIdx.x & 31;
  if (node >= NNODE) return;
  float d = dists[(size_t)node * KNN_ + lane];
  float bias = rbp[0];
#pragma unroll
  for (int r = 0; r < NRBF_; ++r) {
    float t = d - centers[r];
    bias += __expf(-RBF_GAMMA_ * t * t) * rw[r];
  }
  float m = bias;
#pragma unroll
  for (int o = 16; o > 0; o >>= 1) m = fmaxf(m, __shfl_xor(m, o, 32));
  float e = __expf(bias - m);
  float s = e;
#pragma unroll
  for (int o = 16; o > 0; o >>= 1) s += __shfl_xor(s, o, 32);
  edgew[(size_t)node * KNN_ + lane] = e / s;
}

// ---------------------------------------------------------------------
// Aggregate + concat: one thread per (node, h). Writes hcat[n] =
// [nodes_f16(64) | f16(sum_k w_k * nodes[sender_k][h]) (64)].
// ---------------------------------------------------------------------
__global__ void agg_hcat_kernel(const float* __restrict__ edgew,
                                const int* __restrict__ senders,
                                const float* __restrict__ nodes,
                                const _Float16* __restrict__ nodesH,
                                _Float16* __restrict__ hcat) {
  int t = blockIdx.x * 256 + threadIdx.x;   // NNODE*64 threads
  int h = t & (H_ - 1);
  int n = t >> 6;
  if (n >= NNODE) return;
  float acc = 0.0f;
#pragma unroll 8
  for (int k = 0; k < KNN_; ++k) {
    int s = senders[(size_t)n * KNN_ + k];
    acc += edgew[(size_t)n * KNN_ + k] * nodes[(size_t)s * H_ + h];
  }
  hcat[(size_t)n * 128 + h]       = nodesH[(size_t)n * H_ + h];
  hcat[(size_t)n * 128 + 64 + h]  = (_Float16)acc;
}

// ---------------------------------------------------------------------
// Head final layer 64->2 + softplus: one thread per test node.
// d_out layout: mean[16384] then std[16384].
// ---------------------------------------------------------------------
__global__ void head_final_kernel(const _Float16* __restrict__ hb,
                                  const float* __restrict__ w2,
                                  const float* __restrict__ b2,
                                  float* __restrict__ out) {
  int m = blockIdx.x * 256 + threadIdx.x;
  if (m >= NTEST) return;
  float o0 = b2[0], o1 = b2[1];
#pragma unroll
  for (int i = 0; i < H_; ++i) {
    float v = (float)hb[(size_t)m * H_ + i];
    o0 += v * w2[i * 2 + 0];
    o1 += v * w2[i * 2 + 1];
  }
  float sp = (o1 > 20.0f) ? o1 : log1pf(expf(o1));
  out[m]          = o0;
  out[NTEST + m]  = sp + 1e-3f;
}

// =====================================================================
extern "C" void kernel_launch(void* const* d_in, const int* in_sizes, int n_in,
                              void* d_out, int out_size, void* d_ws, size_t ws_size,
                              hipStream_t stream) {
  (void)in_sizes; (void)n_in; (void)out_size; (void)ws_size;

  const float* s_ctx      = (const float*)d_in[0];
  const float* f_ctx      = (const float*)d_in[1];
  const float* s_test     = (const float*)d_in[2];
  const float* obs_table  = (const float*)d_in[3];
  const float* emb_w0     = (const float*)d_in[4];
  const float* emb_b0     = (const float*)d_in[5];
  const float* emb_w1     = (const float*)d_in[6];
  const float* emb_b1     = (const float*)d_in[7];
  const float* emb_w2     = (const float*)d_in[8];
  const float* emb_b2     = (const float*)d_in[9];
  const float* ln_g       = (const float*)d_in[10];
  const float* ln_b       = (const float*)d_in[11];
  const float* rbf_centers= (const float*)d_in[12];
  const float* rbf_w      = (const float*)d_in[13];
  const float* rbf_b      = (const float*)d_in[14];
  const float* blk_w1     = (const float*)d_in[15];
  const float* blk_b1     = (const float*)d_in[16];
  const float* blk_w2     = (const float*)d_in[17];
  const float* blk_b2     = (const float*)d_in[18];
  const float* blk_ln_g   = (const float*)d_in[19];
  const float* blk_ln_b   = (const float*)d_in[20];
  const float* head_w0    = (const float*)d_in[21];
  const float* head_b0    = (const float*)d_in[22];
  const float* head_w1    = (const float*)d_in[23];
  const float* head_b1    = (const float*)d_in[24];
  const float* head_w2    = (const float*)d_in[25];
  const float* head_b2    = (const float*)d_in[26];

  // ---------------- workspace carving (256B aligned) ----------------
  char* ws = (char*)d_ws;
  size_t o = 0;
  auto carve = [&](size_t bytes) -> char* {
    char* p = ws + o;
    o = (o + bytes + 255) & ~(size_t)255;
    return p;
  };
  _Float16* pw1 = (_Float16*)carve((size_t)256 * 128 * 2);
  _Float16* pw2 = (_Float16*)carve((size_t)128 * 64 * 2);
  _Float16* pb1 = (_Float16*)carve((size_t)NBLK_ * 128 * 128 * 2);
  _Float16* pb2 = (_Float16*)carve((size_t)NBLK_ * 128 * 64 * 2);
  _Float16* ph0 = (_Float16*)carve((size_t)64 * 256 * 2);
  _Float16* ph1 = (_Float16*)carve((size_t)256 * 64 * 2);
  _Float16* bufA   = (_Float16*)carve((size_t)NNODE * 256 * 2); // h0 / t1 / head hA
  _Float16* bufB   = (_Float16*)carve((size_t)NNODE * 128 * 2); // h1 / hcat / head hB
  float*    bufF   = (float*)   carve((size_t)NNODE * H_ * 4);  // embed-out / t2
  float*    nodes  = (float*)   carve((size_t)NNODE * H_ * 4);
  _Float16* nodesH = (_Float16*)carve((size_t)NNODE * H_ * 2);
  int*      senders= (int*)     carve((size_t)NNODE * KNN_ * 4);
  float*    dists  = (float*)   carve((size_t)NNODE * KNN_ * 4);
  float*    edgew  = (float*)   carve((size_t)NNODE * KNN_ * 4);

  // ---------------- pack all WMMA weights to f16 fragments ----------
  auto pack = [&](const float* W, _Float16* out, int Kd, int N) {
    int total = Kd * N;
    pack_w_kernel<<<(total + 255) / 256, 256, 0, stream>>>(W, out, Kd, N);
  };
  pack(emb_w1, pw1, 256, 128);
  pack(emb_w2, pw2, 128, 64);
  for (int l = 0; l < NBLK_; ++l) pack(blk_w1 + (size_t)l * 128 * 128, pb1 + (size_t)l * 128 * 128, 128, 128);
  for (int l = 0; l < NBLK_; ++l) pack(blk_w2 + (size_t)l * 128 * 64,  pb2 + (size_t)l * 128 * 64,  128, 64);
  pack(head_w0, ph0, 64, 256);
  pack(head_w1, ph1, 256, 64);

  // grid for gemm_wmma_v2: (M/64) * (N/64) blocks of 128 threads
  #define GEMM_GRID(M, N) (((M) >> 6) * ((N) >> 6))

  // ---------------- embed MLP + LN ----------------------------------
  embed_l0_kernel<<<NNODE, 256, 0, stream>>>(s_ctx, f_ctx, s_test, obs_table,
                                             emb_w0, emb_b0, bufA);
  gemm_wmma_v2<8, 1><<<GEMM_GRID(NNODE, 128), 128, 0, stream>>>(
      bufA, pw1, emb_b1, bufB, nullptr, NNODE, 128);
  gemm_wmma_v2<4, 0><<<GEMM_GRID(NNODE, 64), 128, 0, stream>>>(
      bufB, pw2, emb_b2, nullptr, bufF, NNODE, 64);
  residual_ln_kernel<<<(NNODE + 255) / 256, 256, 0, stream>>>(
      bufF, nullptr, ln_g, ln_b, nodes, nodesH, NNODE);

  // ---------------- kNN graph ---------------------------------------
  knn_kernel<<<(NNODE + 255) / 256, 256, 0, stream>>>(s_ctx, s_test, senders, dists);

  // ---------------- message-passing blocks --------------------------
  for (int l = 0; l < NBLK_; ++l) {
    edge_softmax_kernel<<<NNODE / 4, 128, 0, stream>>>(
        dists, rbf_centers + l * NRBF_, rbf_w + l * NRBF_, rbf_b + l, edgew);
    agg_hcat_kernel<<<(NNODE * H_) / 256, 256, 0, stream>>>(
        edgew, senders, nodes, nodesH, bufB);
    gemm_wmma_v2<4, 1><<<GEMM_GRID(NNODE, 128), 128, 0, stream>>>(
        bufB, pb1 + (size_t)l * 128 * 128, blk_b1 + l * 128, bufA, nullptr,
        NNODE, 128);
    gemm_wmma_v2<4, 0><<<GEMM_GRID(NNODE, 64), 128, 0, stream>>>(
        bufA, pb2 + (size_t)l * 128 * 64, blk_b2 + l * 64, nullptr, bufF,
        NNODE, 64);
    residual_ln_kernel<<<(NNODE + 255) / 256, 256, 0, stream>>>(
        nodes, bufF, blk_ln_g + l * H_, blk_ln_b + l * H_, nodes, nodesH, NNODE);
  }

  // ---------------- head on test nodes ------------------------------
  gemm_wmma_v2<2, 1><<<GEMM_GRID(NTEST, 256), 128, 0, stream>>>(
      nodesH + (size_t)NCTX * H_, ph0, head_b0, bufA, nullptr, NTEST, 256);
  gemm_wmma_v2<8, 1><<<GEMM_GRID(NTEST, 64), 128, 0, stream>>>(
      bufA, ph1, head_b1, bufB, nullptr, NTEST, 64);
  head_final_kernel<<<(NTEST + 255) / 256, 256, 0, stream>>>(
      bufB, head_w2, head_b2, (float*)d_out);
}
